// GraphNet_61486751809588
// MI455X (gfx1250) — matrix operand order repaired and checked by reading, not compiled, not run on hardware
//
#include <hip/hip_runtime.h>
#include <hip/hip_bf16.h>

#define N_NODES 50000
#define N_EDGES 1000000
#define M_EDGES (2 * N_EDGES + N_NODES)   // 2,050,000 ; multiple of 16
#define HID 64
#define NT_MP (M_EDGES / 16)              // 128125 wave-tiles
#define NT_SC (N_EDGES / 16)              // 62500 wave-tiles

typedef __attribute__((ext_vector_type(16))) __bf16 bf16x16;
typedef __attribute__((ext_vector_type(8)))  float  f32x8;

// ---------- bf16 helpers (bit-level, RNE) ----------
__device__ __forceinline__ unsigned short f2bf(float f) {
  unsigned u = __float_as_uint(f);
  u += 0x7FFFu + ((u >> 16) & 1u);
  return (unsigned short)(u >> 16);
}
__device__ __forceinline__ float bf2f(unsigned short h) {
  return __uint_as_float(((unsigned)h) << 16);
}

// ---------- WMMA fragment loaders ----------
// A tile 16x32 bf16 from row-major [16 x 128] bf16 buffer (LDS).
// ISA layout: lanes 0-15 hold row M=lane, K = {0..7, 16..23}; lanes 16-31 K = {8..15, 24..31}.
__device__ __forceinline__ bf16x16 load_a(const unsigned short* buf, int lane, int kblk) {
  int row = lane & 15;
  int ks  = (lane >> 4) << 3;                       // 0 or 8
  const unsigned short* p = buf + row * 128 + kblk * 32 + ks;
  union { bf16x16 v; uint4 q[2]; } u;
  u.q[0] = *(const uint4*)(p);                      // K ks..ks+7
  u.q[1] = *(const uint4*)(p + 16);                 // K ks+16..ks+23
  return u.v;
}
// B tile 32x16 bf16 from pre-packed tiles: tile*32 lanes * 16 contiguous K values.
__device__ __forceinline__ bf16x16 load_b(const unsigned short* tiles, int tileIdx, int lane) {
  const unsigned short* p = tiles + ((size_t)(tileIdx * 32 + lane)) * 16;
  union { bf16x16 v; uint4 q[2]; } u;
  u.q[0] = *(const uint4*)(p);
  u.q[1] = *(const uint4*)(p + 8);
  return u.v;
}
__device__ __forceinline__ f32x8 wmma_bf16(bf16x16 a, bf16x16 b, f32x8 c) {
  return __builtin_amdgcn_wmma_f32_16x16x32_bf16(false, a, false, b, (short)0, c, false, false);
}

// ---------- weight repack: f32 [K x N] row-major -> bf16 WMMA B tiles ----------
__global__ void pack_w(const float* __restrict__ src, unsigned short* __restrict__ dst,
                       int K, int N) {
  int id = blockIdx.x * blockDim.x + threadIdx.x;
  if (id >= K * N) return;
  int kk = id & 15;
  int l  = (id >> 4) & 31;
  int t  = id >> 9;                // tile index, 512 elems per tile
  int NT = N >> 4;
  int nt = t % NT;
  int kt = t / NT;
  int krow = kt * 32 + ((kk >> 3) << 4) + ((l >> 4) << 3) + (kk & 7);
  int ncol = nt * 16 + (l & 15);
  dst[id] = f2bf(src[krow * N + ncol]);
}

// ---------- input network: H = relu(tanh(BN(x@W1+b1)) @ W2 + b2), stored bf16 ----------
__global__ __launch_bounds__(256) void input_net(
    const float* __restrict__ x, const float* __restrict__ W1, const float* __restrict__ b1,
    const float* __restrict__ gamma, const float* __restrict__ beta,
    const float* __restrict__ mean, const float* __restrict__ var,
    const float* __restrict__ W2, const float* __restrict__ b2,
    unsigned short* __restrict__ Hbf) {
  __shared__ float sW2[64 * 64];
  __shared__ float sW1[8 * 64];
  __shared__ float sh[4][64];
  int t = threadIdx.x;
  for (int i = t; i < 64 * 64; i += 256) sW2[i] = W2[i];
  for (int i = t; i < 8 * 64; i += 256) sW1[i] = W1[i];
  __syncthreads();
  int j = t & 63, ln = t >> 6;
  int node = blockIdx.x * 4 + ln;
  if (node < N_NODES) {
    float h = b1[j];
#pragma unroll
    for (int k = 0; k < 8; ++k) h += x[node * 8 + k] * sW1[k * 64 + j];
    h = (h - mean[j]) * rsqrtf(var[j] + 1e-5f) * gamma[j] + beta[j];
    sh[ln][j] = tanhf(h);
  }
  __syncthreads();
  if (node < N_NODES) {
    float v = b2[j];
#pragma unroll 8
    for (int k = 0; k < 64; ++k) v += sh[ln][k] * sW2[k * 64 + j];
    Hbf[(size_t)node * 64 + j] = f2bf(fmaxf(v, 0.f));
  }
}

__global__ void zero_f32(float* __restrict__ p, int n) {
  for (int i = blockIdx.x * blockDim.x + threadIdx.x; i < n; i += gridDim.x * blockDim.x)
    p[i] = 0.f;
}
__global__ void cvt_bf(const float* __restrict__ s, unsigned short* __restrict__ d, int n) {
  for (int i = blockIdx.x * blockDim.x + threadIdx.x; i < n; i += gridDim.x * blockDim.x)
    d[i] = f2bf(s[i]);
}

// mp edge list (concat fwd / reversed / self-loops), computed on the fly
__device__ __forceinline__ void mp_edge(int m, const int* __restrict__ es,
                                        const int* __restrict__ ed, int& s, int& d) {
  if (m < N_EDGES)          { s = es[m];            d = ed[m]; }
  else if (m < 2 * N_EDGES) { s = ed[m - N_EDGES];  d = es[m - N_EDGES]; }
  else                      { s = m - 2 * N_EDGES;  d = s; }
}

// ---------- message passing: per 16-edge tile, relu(relu([xi, xj-xi]@Wc1+bc1)@Wc2+bc2) scatter-add ----------
__global__ __launch_bounds__(128) void mp_pass(
    const unsigned short* __restrict__ Hbf, float* __restrict__ Hnext,
    const int* __restrict__ es, const int* __restrict__ ed,
    const unsigned short* __restrict__ wc1t, const unsigned short* __restrict__ wc2t,
    const float* __restrict__ bc1, const float* __restrict__ bc2) {
  __shared__ unsigned short s_wc1[32 * 32 * 16];    // 32 tiles (kb*8+nt), 32KB
  __shared__ float s_bc1[128];
  __shared__ float s_bc2[64];
  __shared__ unsigned short s_msg[4][16 * 128];     // per-wave A staging, 4x4KB
  __shared__ int s_dst[4][16];
  int t = threadIdx.x;
  {
    uint4* dv = (uint4*)s_wc1;
    const uint4* sv = (const uint4*)wc1t;
    for (int i = t; i < 2048; i += 128) dv[i] = sv[i];
  }
  if (t < 128) s_bc1[t] = bc1[t];
  if (t < 64)  s_bc2[t] = bc2[t];
  __syncthreads();

  int wave = t >> 5, lane = t & 31;
  int tile = blockIdx.x * 4 + wave;
  if (tile >= NT_MP) return;
  int mbase = tile * 16;

  // ---- gather: lane pair per edge; row = [x_i , x_j - x_i] in bf16 ----
  int e = lane >> 1, h = lane & 1;
  int m = mbase + e;
  int s, d;
  mp_edge(m, es, ed, s, d);
  if (h == 0) s_dst[wave][e] = d;
  const uint4* ps = (const uint4*)(Hbf + (size_t)s * 64 + h * 32);
  const uint4* pd = (const uint4*)(Hbf + (size_t)d * 64 + h * 32);
  uint4 qs[4], qd[4];
#pragma unroll
  for (int i = 0; i < 4; ++i) { qs[i] = ps[i]; qd[i] = pd[i]; }
  unsigned short* row = s_msg[wave] + e * 128;
  uint4* xi = (uint4*)(row + h * 32);
  uint4* xj = (uint4*)(row + 64 + h * 32);
#pragma unroll
  for (int i = 0; i < 4; ++i) {
    xi[i] = qd[i];                                  // x_i = H[dst]
    uint4 dq;
    unsigned* du = (unsigned*)&dq;
    const unsigned* su = (const unsigned*)&qs[i];
    const unsigned* tu = (const unsigned*)&qd[i];
#pragma unroll
    for (int w = 0; w < 4; ++w) {
      float a0 = bf2f((unsigned short)(su[w] & 0xFFFFu)) - bf2f((unsigned short)(tu[w] & 0xFFFFu));
      float a1 = bf2f((unsigned short)(su[w] >> 16))     - bf2f((unsigned short)(tu[w] >> 16));
      du[w] = (unsigned)f2bf(a0) | ((unsigned)f2bf(a1) << 16);
    }
    xj[i] = dq;                                     // x_j - x_i
  }
  __builtin_amdgcn_wave_barrier();

  // ---- layer 1: [16x128] @ [128x128] ----
  f32x8 acc[8] = {};
#pragma unroll
  for (int kb = 0; kb < 4; ++kb) {
    bf16x16 a = load_a(s_msg[wave], lane, kb);
#pragma unroll
    for (int nt = 0; nt < 8; ++nt) {
      bf16x16 b = load_b(s_wc1, kb * 8 + nt, lane);
      acc[nt] = wmma_bf16(a, b, acc[nt]);
    }
  }
  __builtin_amdgcn_wave_barrier();

  // ---- bias + relu, restage as bf16 A tile (LDS is in-order; wave-private buffer) ----
  int n = lane & 15, mb = (lane >> 4) << 3;
#pragma unroll
  for (int nt = 0; nt < 8; ++nt) {
    float bias = s_bc1[nt * 16 + n];
#pragma unroll
    for (int i = 0; i < 8; ++i) {
      float v = fmaxf(acc[nt][i] + bias, 0.f);
      s_msg[wave][(mb + i) * 128 + nt * 16 + n] = f2bf(v);
    }
  }
  __builtin_amdgcn_wave_barrier();

  // ---- layer 2: [16x128] @ [128x64], B streamed from global (L2/WGP$-resident) ----
  f32x8 acc2[4] = {};
#pragma unroll
  for (int kb = 0; kb < 4; ++kb) {
    bf16x16 a = load_a(s_msg[wave], lane, kb);
#pragma unroll
    for (int nt = 0; nt < 4; ++nt) {
      bf16x16 b = load_b(wc2t, kb * 4 + nt, lane);
      acc2[nt] = wmma_bf16(a, b, acc2[nt]);
    }
  }

  // ---- bias + relu + segment-sum scatter (native f32 atomics) ----
#pragma unroll
  for (int nt = 0; nt < 4; ++nt) {
    float bias = s_bc2[nt * 16 + n];
#pragma unroll
    for (int i = 0; i < 8; ++i) {
      float v = fmaxf(acc2[nt][i] + bias, 0.f);
      int drow = s_dst[wave][mb + i];
      unsafeAtomicAdd(&Hnext[(size_t)drow * 64 + nt * 16 + n], v);
    }
  }
}

// ---------- edge scorer: sigmoid(relu([H_src,H_dst]@We1+be1) @ We2 + be2) ----------
__global__ __launch_bounds__(128) void edge_score(
    const unsigned short* __restrict__ Hbf,
    const int* __restrict__ es, const int* __restrict__ ed,
    const unsigned short* __restrict__ we1t,
    const float* __restrict__ be1, const float* __restrict__ We2,
    const float* __restrict__ be2, float* __restrict__ out) {
  __shared__ unsigned short s_we1[16 * 32 * 16];    // 16 tiles, 16KB
  __shared__ float s_be1[64], s_we2[64];
  __shared__ unsigned short s_msg[4][16 * 128];
  int t = threadIdx.x;
  {
    uint4* dv = (uint4*)s_we1;
    const uint4* sv = (const uint4*)we1t;
    for (int i = t; i < 1024; i += 128) dv[i] = sv[i];
  }
  if (t < 64) { s_be1[t] = be1[t]; s_we2[t] = We2[t]; }
  __syncthreads();

  int wave = t >> 5, lane = t & 31;
  int tile = blockIdx.x * 4 + wave;
  if (tile >= NT_SC) return;
  int mbase = tile * 16;

  int e = lane >> 1, h = lane & 1;
  int m = mbase + e;
  int s = es[m], d = ed[m];
  const uint4* ps = (const uint4*)(Hbf + (size_t)s * 64 + h * 32);
  const uint4* pd = (const uint4*)(Hbf + (size_t)d * 64 + h * 32);
  unsigned short* row = s_msg[wave] + e * 128;
  uint4* c0 = (uint4*)(row + h * 32);
  uint4* c1 = (uint4*)(row + 64 + h * 32);
#pragma unroll
  for (int i = 0; i < 4; ++i) { c0[i] = ps[i]; c1[i] = pd[i]; }
  __builtin_amdgcn_wave_barrier();

  f32x8 acc[4] = {};
#pragma unroll
  for (int kb = 0; kb < 4; ++kb) {
    bf16x16 a = load_a(s_msg[wave], lane, kb);
#pragma unroll
    for (int nt = 0; nt < 4; ++nt) {
      bf16x16 b = load_b(s_we1, kb * 4 + nt, lane);
      acc[nt] = wmma_bf16(a, b, acc[nt]);
    }
  }
  __builtin_amdgcn_wave_barrier();

  // stage relu(layer1) rows [16 x 64] bf16 (reuse buffer)
  int n = lane & 15, mb = (lane >> 4) << 3;
#pragma unroll
  for (int nt = 0; nt < 4; ++nt) {
    float bias = s_be1[nt * 16 + n];
#pragma unroll
    for (int i = 0; i < 8; ++i) {
      float v = fmaxf(acc[nt][i] + bias, 0.f);
      s_msg[wave][(mb + i) * 64 + nt * 16 + n] = f2bf(v);
    }
  }
  __builtin_amdgcn_wave_barrier();

  // final 64->1 dot: lane pair per edge, shuffle-combine, sigmoid
  float sum = 0.f;
  const unsigned short* r = s_msg[wave] + e * 64 + h * 32;
#pragma unroll
  for (int k = 0; k < 32; ++k) sum += bf2f(r[k]) * s_we2[h * 32 + k];
  sum += __shfl_xor(sum, 1, 32);
  if (h == 0) {
    float z = sum + be2[0];
    out[m] = 1.f / (1.f + __expf(-z));
  }
}

// ---------- host orchestration ----------
extern "C" void kernel_launch(void* const* d_in, const int* in_sizes, int n_in,
                              void* d_out, int out_size, void* d_ws, size_t ws_size,
                              hipStream_t stream) {
  const float* x       = (const float*)d_in[0];
  const int*   es      = (const int*)d_in[1];
  const int*   ed      = (const int*)d_in[2];
  const float* W1      = (const float*)d_in[3];
  const float* b1      = (const float*)d_in[4];
  const float* gamma   = (const float*)d_in[5];
  const float* beta    = (const float*)d_in[6];
  const float* bn_mean = (const float*)d_in[7];
  const float* bn_var  = (const float*)d_in[8];
  const float* W2      = (const float*)d_in[9];
  const float* b2      = (const float*)d_in[10];
  const float* Wc1     = (const float*)d_in[11];
  const float* bc1     = (const float*)d_in[12];
  const float* Wc2     = (const float*)d_in[13];
  const float* bc2     = (const float*)d_in[14];
  const float* We1     = (const float*)d_in[15];
  const float* be1     = (const float*)d_in[16];
  const float* We2     = (const float*)d_in[17];
  const float* be2     = (const float*)d_in[18];

  char* ws = (char*)d_ws;
  unsigned short* Hbf   = (unsigned short*)(ws);                 //  6,400,000 B
  float*          Hnext = (float*)(ws + 6400000);                // 12,800,000 B
  unsigned short* wc1t  = (unsigned short*)(ws + 19200000);      //     32,768 B
  unsigned short* wc2t  = (unsigned short*)(ws + 19232768);      //     16,384 B
  unsigned short* we1t  = (unsigned short*)(ws + 19249152);      //     16,384 B

  pack_w<<<(128 * 128 + 255) / 256, 256, 0, stream>>>(Wc1, wc1t, 128, 128);
  pack_w<<<(128 * 64 + 255) / 256, 256, 0, stream>>>(Wc2, wc2t, 128, 64);
  pack_w<<<(128 * 64 + 255) / 256, 256, 0, stream>>>(We1, we1t, 128, 64);

  input_net<<<N_NODES / 4, 256, 0, stream>>>(x, W1, b1, gamma, beta, bn_mean, bn_var,
                                             W2, b2, Hbf);

  for (int it = 0; it < 3; ++it) {
    zero_f32<<<2048, 256, 0, stream>>>(Hnext, N_NODES * HID);
    mp_pass<<<(NT_MP + 3) / 4, 128, 0, stream>>>(Hbf, Hnext, es, ed, wc1t, wc2t, bc1, bc2);
    cvt_bf<<<2048, 256, 0, stream>>>(Hnext, Hbf, N_NODES * HID);
  }

  edge_score<<<(NT_SC + 3) / 4, 128, 0, stream>>>(Hbf, es, ed, we1t, be1, We2, be2,
                                                  (float*)d_out);
}
